// UnfoldConv3d_6270652252407
// MI455X (gfx1250) — matrix operand order repaired and checked by reading, not compiled, not run on hardware
//
#include <hip/hip_runtime.h>

typedef __attribute__((ext_vector_type(16))) __bf16 v16bf;
typedef __attribute__((ext_vector_type(8)))  float  v8f;

#define PD      1536   // patch dim (K)
#define OC      1024   // out channels (N)
#define LDSTR   1544   // padded LDS row stride (bf16 elems)
#define NPATCH  25088  // 16 * 8 * 14 * 14
#define MTILE   32     // patches per workgroup
#define EPSV    1e-5f

__device__ __forceinline__ unsigned short f2bf(float f) {
  union { float f; unsigned u; } v; v.f = f;
  unsigned r = v.u + 0x7FFFu + ((v.u >> 16) & 1u);   // round-to-nearest-even
  return (unsigned short)(r >> 16);
}

union Frag { v16bf v; float4 f4[2]; };

// ---------------------------------------------------------------------------
// Prep: Wg = bf16(W * gamma), G[n] = sum_k gamma[k]*W[n,k],
//       cbias[n] = b_lin[n] + sum_k beta[k]*W[n,k]
// ---------------------------------------------------------------------------
__global__ __launch_bounds__(256) void prep_kernel(
    const float* __restrict__ w_lin, const float* __restrict__ gamma,
    const float* __restrict__ beta,  const float* __restrict__ b_lin,
    unsigned short* __restrict__ wg, float* __restrict__ G,
    float* __restrict__ cbias) {
  __shared__ float redG[8], redB[8];
  const int n = blockIdx.x, t = threadIdx.x;
  const float* wrow = w_lin + (size_t)n * PD;
  float sG = 0.f, sB = 0.f;
  for (int k = t; k < PD; k += 256) {
    float w   = wrow[k];
    float wgv = w * gamma[k];
    wg[(size_t)n * PD + k] = f2bf(wgv);
    sG += wgv;
    sB += beta[k] * w;
  }
#pragma unroll
  for (int m = 16; m >= 1; m >>= 1) {
    sG += __shfl_xor(sG, m, 32);
    sB += __shfl_xor(sB, m, 32);
  }
  if ((t & 31) == 0) { redG[t >> 5] = sG; redB[t >> 5] = sB; }
  __syncthreads();
  if (t == 0) {
    float g = 0.f, bb = 0.f;
#pragma unroll
    for (int i = 0; i < 8; ++i) { g += redG[i]; bb += redB[i]; }
    G[n]     = g;
    cbias[n] = b_lin[n] + bb;
  }
}

// ---------------------------------------------------------------------------
// Fused: gather 32 patches -> LDS(bf16) + LN stats, then 32x1024 GEMM (K=1536)
// with v_wmma_f32_16x16x32_bf16 (each B fragment feeds 2 WMMAs), LN folded
// into the epilogue.
// ---------------------------------------------------------------------------
__global__ __launch_bounds__(256) void fused_patch_ln_gemm(
    const float* __restrict__ x, const unsigned short* __restrict__ wg,
    const float* __restrict__ G, const float* __restrict__ cbias,
    float* __restrict__ out) {
  __shared__ __align__(16) unsigned short sA[MTILE * LDSTR];  // ~98.8 KB
  __shared__ float sMu[MTILE], sRstd[MTILE];

  const int tid = threadIdx.x;
  const int P0  = blockIdx.x * MTILE;

  // ---- Stage 1: gather + stats (two 16-patch halves) ----------------------
#pragma unroll
  for (int half = 0; half < 2; ++half) {
    const int p  = half * 16 + (tid >> 4);  // patch within tile
    const int kw = tid & 15;                // contiguous W within patch
    const int P  = P0 + p;
    const int b  = P / 1568;                // 1568 = 8 * 196
    const int rm = P % 1568;
    const int tt = rm / 196;
    const int hw = rm % 196;
    const int h  = hw / 14, w = hw % 14;
    // x strides: B:3*16*224*224  C:16*224*224  T:224*224  H:224  W:1
    const float* xb = x + (size_t)b * 2408448 + (size_t)(2 * tt) * 50176
                        + (size_t)(16 * h) * 224 + 16 * w + kw;
    float sum = 0.f, ssq = 0.f;
    unsigned short* srow = sA + p * LDSTR;
#pragma unroll
    for (int g = 0; g < 96; ++g) {          // g = (c*2+kt)*16+kh ; f = g*16+kw
      const int c = g >> 5, kt = (g >> 4) & 1, kh = g & 15;
      float v = __builtin_nontemporal_load(
                    xb + (size_t)c * 802816 + (size_t)kt * 50176 + kh * 224);
      sum += v; ssq += v * v;
      srow[g * 16 + kw] = f2bf(v);
    }
#pragma unroll
    for (int m = 8; m >= 1; m >>= 1) {      // reduce within 16-lane groups
      sum += __shfl_xor(sum, m, 32);
      ssq += __shfl_xor(ssq, m, 32);
    }
    if (kw == 0) {
      float mu  = sum * (1.f / PD);
      float var = ssq * (1.f / PD) - mu * mu;
      sMu[p]   = mu;
      sRstd[p] = rsqrtf(var + EPSV);
    }
  }
  __syncthreads();

  // ---- Stage 2: WMMA GEMM (M=32 via two halves per wave) ------------------
  const int wave = tid >> 5, lane = tid & 31;
  const int l16  = lane & 15;
  const int hi   = (lane >> 4) & 1;

  v8f acc[16];
#pragma unroll
  for (int i = 0; i < 16; ++i) acc[i] = (v8f)0.f;

  // A operand, this code's contiguous k order: lane<16 -> K 0..15; lane>=16 -> K 16..31
  const unsigned short* aBase0 = sA + l16 * LDSTR + (hi ? 16 : 0);
  const unsigned short* aBase1 = aBase0 + 16 * LDSTR;
  // B operand, same k order as A: lane = N, 16 contiguous K; lanes>=16 -> K+16
  const unsigned short* brow[8];
#pragma unroll
  for (int i = 0; i < 8; ++i)
    brow[i] = wg + (size_t)((wave * 8 + i) * 16 + l16) * PD + (hi ? 16 : 0);

  // 2-deep B double buffer; prefetch rotates through the unrolled tile loop
  Frag bb[2];
  bb[0].f4[0] = *(const float4*)(brow[0]);
  bb[0].f4[1] = *(const float4*)(brow[0] + 8);

  for (int k0 = 0; k0 < PD; k0 += 32) {
    Frag a0, a1;
    a0.f4[0] = *(const float4*)(aBase0 + k0);
    a0.f4[1] = *(const float4*)(aBase0 + k0 + 8);
    a1.f4[0] = *(const float4*)(aBase1 + k0);
    a1.f4[1] = *(const float4*)(aBase1 + k0 + 8);
#pragma unroll
    for (int i = 0; i < 8; ++i) {
      // prefetch next B tile (wraps to next K-step; final overread stays
      // inside the workspace allocation and is never consumed)
      const unsigned short* bp =
          (i < 7) ? (brow[i + 1] + k0) : (brow[0] + k0 + 32);
      bb[(i + 1) & 1].f4[0] = *(const float4*)(bp);
      bb[(i + 1) & 1].f4[1] = *(const float4*)(bp + 8);
      acc[i]     = __builtin_amdgcn_wmma_f32_16x16x32_bf16(
                       false, a0.v, false, bb[i & 1].v, (short)0, acc[i],
                       false, false);
      acc[8 + i] = __builtin_amdgcn_wmma_f32_16x16x32_bf16(
                       false, a1.v, false, bb[i & 1].v, (short)0, acc[8 + i],
                       false, false);
    }
  }

  // ---- Epilogue: out = rstd*acc - mu*rstd*G[n] + cbias[n] -----------------
  const int mlow = hi ? 8 : 0;     // C layout: VGPR r -> M = r + 8*hi, N = l16
#pragma unroll
  for (int half = 0; half < 2; ++half) {
    const int mbase = half * 16 + mlow;
    float rs[8], ms[8];
#pragma unroll
    for (int r = 0; r < 8; ++r) {
      rs[r] = sRstd[mbase + r];
      ms[r] = sMu[mbase + r] * rs[r];
    }
#pragma unroll
    for (int i = 0; i < 8; ++i) {
      const int n  = (wave * 8 + i) * 16 + l16;
      const float Gn = G[n], cb = cbias[n];
#pragma unroll
      for (int r = 0; r < 8; ++r)
        __builtin_nontemporal_store(
            rs[r] * acc[half * 8 + i][r] - ms[r] * Gn + cb,
            out + (size_t)(P0 + mbase + r) * OC + n);
    }
  }
}

// ---------------------------------------------------------------------------
extern "C" void kernel_launch(void* const* d_in, const int* in_sizes, int n_in,
                              void* d_out, int out_size, void* d_ws, size_t ws_size,
                              hipStream_t stream) {
  const float* x     = (const float*)d_in[0];
  const float* gamma = (const float*)d_in[1];
  const float* beta  = (const float*)d_in[2];
  const float* w_lin = (const float*)d_in[3];
  const float* b_lin = (const float*)d_in[4];
  float* out = (float*)d_out;

  unsigned short* wg = (unsigned short*)d_ws;                       // 3 MB bf16 W*gamma
  float* G     = (float*)((char*)d_ws + (size_t)OC * PD * sizeof(unsigned short));
  float* cbias = G + OC;

  prep_kernel<<<OC, 256, 0, stream>>>(w_lin, gamma, beta, b_lin, wg, G, cbias);
  fused_patch_ln_gemm<<<NPATCH / MTILE, 256, 0, stream>>>(x, wg, G, cbias, out);
}
